// SwinTransformer_28398323761710
// MI455X (gfx1250) — matrix operand order repaired and checked by reading, not compiled, not run on hardware
//
#include <hip/hip_runtime.h>
#include <hip/hip_fp16.h>

typedef __attribute__((ext_vector_type(16))) _Float16 v16h;
typedef __attribute__((ext_vector_type(8)))  float    v8f;
typedef _Float16 half_t;

#define BB      32
#define HH      56
#define WW      56
#define CC      256
#define NHEADS  8
#define WSZ     7
#define NWIN    49          // WSZ*WSZ
#define SHIFT_  3
#define HIDDEN_ 1024
#define LL      (HH*WW)     // 3136
#define MROWS   (BB*LL)     // 100352
#define WINB    (BB*64)     // 2048 windows total
#define MMERGE  (BB*784)    // 25088

// ---------------------------------------------------------------------------
// helpers
// ---------------------------------------------------------------------------
__device__ __forceinline__ int win_src_row(int r, int shift) {
    // windowed row r = win*49 + n  ->  row index in the flat [B, H*W] image
    int win = r / NWIN, n = r - win * NWIN;
    int b = win >> 6, wloc = win & 63;
    int wh = wloc >> 3, ww = wloc & 7;
    int i = n / WSZ, j = n - i * WSZ;
    int gh = wh * WSZ + i + shift; if (gh >= HH) gh -= HH;
    int gw = ww * WSZ + j + shift; if (gw >= WW) gw -= WW;
    return b * LL + gh * WW + gw;
}

__device__ __forceinline__ int shift_region(int t) {
    // region id along one axis for the shifted-window attention mask
    return (t < HH - WSZ) ? 0 : ((t < HH - SHIFT_) ? 1 : 2);
}

// ---------------------------------------------------------------------------
// f32 -> f16 weight conversion
// ---------------------------------------------------------------------------
__global__ void cvt_f32_f16(const float* __restrict__ s, half_t* __restrict__ d, int n) {
    int i = blockIdx.x * 256 + threadIdx.x;
    if (i < n) d[i] = (half_t)s[i];
}

// ---------------------------------------------------------------------------
// LayerNorm over C=256, optional shifted-window gather, f16 output
// one block (256 threads) per output row
// ---------------------------------------------------------------------------
__global__ void ln_kernel(const float* __restrict__ x, const float* __restrict__ g,
                          const float* __restrict__ b, half_t* __restrict__ dst,
                          int windowed, int shift) {
    __shared__ float red[256];
    int r = blockIdx.x;
    int src = windowed ? win_src_row(r, shift) : r;
    int t = threadIdx.x;
    float v = x[(size_t)src * CC + t];
    red[t] = v; __syncthreads();
    for (int s = 128; s > 0; s >>= 1) { if (t < s) red[t] += red[t + s]; __syncthreads(); }
    float mu = red[0] * (1.0f / CC);
    __syncthreads();
    float d = v - mu;
    red[t] = d * d; __syncthreads();
    for (int s = 128; s > 0; s >>= 1) { if (t < s) red[t] += red[t + s]; __syncthreads(); }
    float var = red[0] * (1.0f / CC);
    float y = d * rsqrtf(var + 1e-5f) * g[t] + b[t];
    dst[(size_t)r * CC + t] = (half_t)y;
}

// ---------------------------------------------------------------------------
// Patch-merging gather (2x2 neighborhood concat, C'=1024) + LayerNorm, f16 out
// one block (256 threads, 4 channels each) per output row
// ---------------------------------------------------------------------------
__global__ void merge_ln_kernel(const float* __restrict__ x, const float* __restrict__ g,
                                const float* __restrict__ b, half_t* __restrict__ dst) {
    __shared__ float red[256];
    int r = blockIdx.x;                 // 0 .. 25087
    int t = threadIdx.x;
    int bi = r / 784, p = r - bi * 784;
    int oh = p / 28, ow = p - oh * 28;
    float vals[4]; float s = 0.f;
#pragma unroll
    for (int q = 0; q < 4; ++q) {
        int dh = q & 1, dw = q >> 1;    // q: 0=(0,0) 1=(1,0) 2=(0,1) 3=(1,1)
        int srow = bi * LL + (2 * oh + dh) * WW + (2 * ow + dw);
        vals[q] = x[(size_t)srow * CC + t];
        s += vals[q];
    }
    red[t] = s; __syncthreads();
    for (int k = 128; k > 0; k >>= 1) { if (t < k) red[t] += red[t + k]; __syncthreads(); }
    float mu = red[0] * (1.0f / 1024.0f);
    __syncthreads();
    float ss = 0.f;
#pragma unroll
    for (int q = 0; q < 4; ++q) { float d = vals[q] - mu; ss += d * d; }
    red[t] = ss; __syncthreads();
    for (int k = 128; k > 0; k >>= 1) { if (t < k) red[t] += red[t + k]; __syncthreads(); }
    float rs = rsqrtf(red[0] * (1.0f / 1024.0f) + 1e-5f);
#pragma unroll
    for (int q = 0; q < 4; ++q) {
        int c = q * 256 + t;
        dst[(size_t)r * 1024 + c] = (half_t)((vals[q] - mu) * rs * g[c] + b[c]);
    }
}

// ---------------------------------------------------------------------------
// WMMA GEMM:  out = A[M,K](f16,row-major) @ W[N,K](f16,row-major)^T  (+ epilogue)
// block = 128 threads = 4 waves; each wave owns a 32(M) x 64(N) output tile:
// 2 A fragments x 4 B fragments -> 8 x v_wmma_f32_16x16x32_f16 per K-step
// (12 b128 loads / 8 WMMAs; B-fragment traffic amortized over both M tiles).
// ---------------------------------------------------------------------------
enum { EPI_QKV = 0, EPI_PROJ = 1, EPI_GELU = 2, EPI_RES = 3, EPI_PLAIN = 4 };

template <int MODE>
__global__ void gemm_wmma_kernel(const half_t* __restrict__ A,
                                 const half_t* __restrict__ Wt,
                                 const float*  __restrict__ bias,
                                 int Mdim, int Ndim, int Kdim,
                                 half_t* __restrict__ outH,
                                 float*  __restrict__ outF,
                                 const float* __restrict__ resid,
                                 int shift) {
    const int wave = threadIdx.x >> 5;
    const int lane = threadIdx.x & 31;
    const int ntc  = Ndim >> 6;
    int tile = blockIdx.x * 4 + wave;
    int mt = tile / ntc, nc = tile - mt * ntc;
    if (mt >= (Mdim >> 5)) return;
    const int row0 = mt << 5;
    const int hi = lane >> 4;
    const int lr = lane & 15;

    v8f acc0[4] = {};   // rows row0 .. row0+15
    v8f acc1[4] = {};   // rows row0+16 .. row0+31
    // A fragment: lane holds row (lr), 16 contiguous K halves at hi*16
    const half_t* arow0 = A + (size_t)(row0 + lr) * Kdim + (hi ? 16 : 0);
    const half_t* arow1 = arow0 + (size_t)16 * Kdim;
    // B fragment: lane holds column, 16 contiguous K halves at hi*16
    const half_t* bp0 = Wt + (size_t)((nc << 6) + 0 * 16 + lr) * Kdim + (hi ? 16 : 0);
    const half_t* bp1 = Wt + (size_t)((nc << 6) + 1 * 16 + lr) * Kdim + (hi ? 16 : 0);
    const half_t* bp2 = Wt + (size_t)((nc << 6) + 2 * 16 + lr) * Kdim + (hi ? 16 : 0);
    const half_t* bp3 = Wt + (size_t)((nc << 6) + 3 * 16 + lr) * Kdim + (hi ? 16 : 0);

    union U { v16h v; uint4 u[2]; };
    for (int kb = 0; kb < Kdim; kb += 32) {
        U a0, a1;
        a0.u[0] = *reinterpret_cast<const uint4*>(arow0 + kb);
        a0.u[1] = *reinterpret_cast<const uint4*>(arow0 + kb + 8);
        a1.u[0] = *reinterpret_cast<const uint4*>(arow1 + kb);
        a1.u[1] = *reinterpret_cast<const uint4*>(arow1 + kb + 8);
        U w0, w1, w2, w3;
        w0.u[0] = *reinterpret_cast<const uint4*>(bp0 + kb);
        w0.u[1] = *reinterpret_cast<const uint4*>(bp0 + kb + 8);
        w1.u[0] = *reinterpret_cast<const uint4*>(bp1 + kb);
        w1.u[1] = *reinterpret_cast<const uint4*>(bp1 + kb + 8);
        w2.u[0] = *reinterpret_cast<const uint4*>(bp2 + kb);
        w2.u[1] = *reinterpret_cast<const uint4*>(bp2 + kb + 8);
        w3.u[0] = *reinterpret_cast<const uint4*>(bp3 + kb);
        w3.u[1] = *reinterpret_cast<const uint4*>(bp3 + kb + 8);
        acc0[0] = __builtin_amdgcn_wmma_f32_16x16x32_f16(false, a0.v, false, w0.v, (short)0, acc0[0], false, false);
        acc0[1] = __builtin_amdgcn_wmma_f32_16x16x32_f16(false, a0.v, false, w1.v, (short)0, acc0[1], false, false);
        acc0[2] = __builtin_amdgcn_wmma_f32_16x16x32_f16(false, a0.v, false, w2.v, (short)0, acc0[2], false, false);
        acc0[3] = __builtin_amdgcn_wmma_f32_16x16x32_f16(false, a0.v, false, w3.v, (short)0, acc0[3], false, false);
        acc1[0] = __builtin_amdgcn_wmma_f32_16x16x32_f16(false, a1.v, false, w0.v, (short)0, acc1[0], false, false);
        acc1[1] = __builtin_amdgcn_wmma_f32_16x16x32_f16(false, a1.v, false, w1.v, (short)0, acc1[1], false, false);
        acc1[2] = __builtin_amdgcn_wmma_f32_16x16x32_f16(false, a1.v, false, w2.v, (short)0, acc1[2], false, false);
        acc1[3] = __builtin_amdgcn_wmma_f32_16x16x32_f16(false, a1.v, false, w3.v, (short)0, acc1[3], false, false);
    }

    // epilogue: C/D layout -> lane holds column lr, VGPR v = row v + 8*hi
    const int col0 = (nc << 6) + lr;
#pragma unroll
    for (int half = 0; half < 2; ++half) {
        const v8f* accs = half ? acc1 : acc0;
        const int rbase = row0 + half * 16;
#pragma unroll
        for (int sub = 0; sub < 4; ++sub) {
            int col = col0 + sub * 16;
            float bval = 0.f;
            if (MODE != EPI_PLAIN) bval = bias[col];
#pragma unroll
            for (int v = 0; v < 8; ++v) {
                int m = rbase + v + (hi << 3);
                float val = accs[sub][v] + bval;
                if (MODE == EPI_QKV) {
                    if (col < 256) val *= 0.17677669529663687f;   // hd^-0.5 folded into q
                    outH[(size_t)m * Ndim + col] = (half_t)val;
                } else if (MODE == EPI_GELU) {
                    val = 0.5f * val * (1.0f + erff(val * 0.7071067811865476f));
                    outH[(size_t)m * Ndim + col] = (half_t)val;
                } else if (MODE == EPI_PROJ) {
                    int dr = win_src_row(m, shift);               // window-reverse + un-shift
                    outF[(size_t)dr * CC + col] = resid[(size_t)dr * CC + col] + val;
                } else if (MODE == EPI_RES) {
                    outF[(size_t)m * Ndim + col] = resid[(size_t)m * Ndim + col] + val;
                } else {
                    outF[(size_t)m * Ndim + col] = val;
                }
            }
        }
    }
}

// ---------------------------------------------------------------------------
// Window attention, one block (64 threads) per (window, head).
// qkv layout: [wrow, 768] f16 (q already scaled). K/V staged in LDS (f32).
// rel-pos bias index and shift mask computed arithmetically.
// ---------------------------------------------------------------------------
__global__ void attn_kernel(const half_t* __restrict__ qkv, const float* __restrict__ rpb,
                            half_t* __restrict__ out, int masked) {
    __shared__ float Ks[NWIN][32];
    __shared__ float Vs[NWIN][32];
    int blk = blockIdx.x;
    int win = blk >> 3;
    int head = blk & 7;
    int t = threadIdx.x;
    const size_t base = (size_t)win * NWIN * 768;
    const int hoff = head * 32;
    for (int idx = t; idx < NWIN * 32; idx += 64) {
        int m = idx >> 5, d = idx & 31;
        Ks[m][d] = (float)qkv[base + (size_t)m * 768 + 256 + hoff + d];
        Vs[m][d] = (float)qkv[base + (size_t)m * 768 + 512 + hoff + d];
    }
    __syncthreads();
    if (t >= NWIN) return;

    float q[32];
#pragma unroll
    for (int d = 0; d < 32; ++d) q[d] = (float)qkv[base + (size_t)t * 768 + hoff + d];

    int i1 = t / WSZ, j1 = t - i1 * WSZ;
    int wloc = win & 63, wh = wloc >> 3, ww = wloc & 7;
    int reg_n = 0;
    if (masked) reg_n = shift_region(wh * WSZ + i1) * 3 + shift_region(ww * WSZ + j1);

    float sc[NWIN];
    float mx = -1e30f;
    for (int m = 0; m < NWIN; ++m) {
        float s = 0.f;
#pragma unroll
        for (int d = 0; d < 32; ++d) s += q[d] * Ks[m][d];
        int i2 = m / WSZ, j2 = m - i2 * WSZ;
        int rid = (i1 - i2 + WSZ - 1) * (2 * WSZ - 1) + (j1 - j2 + WSZ - 1);
        s += rpb[rid * NHEADS + head];
        if (masked) {
            int reg_m = shift_region(wh * WSZ + i2) * 3 + shift_region(ww * WSZ + j2);
            if (reg_m != reg_n) s -= 100.0f;
        }
        sc[m] = s;
        mx = fmaxf(mx, s);
    }
    float sum = 0.f;
    for (int m = 0; m < NWIN; ++m) { float e = __expf(sc[m] - mx); sc[m] = e; sum += e; }
    float inv = 1.0f / sum;
#pragma unroll
    for (int d = 0; d < 32; ++d) {
        float o = 0.f;
        for (int m = 0; m < NWIN; ++m) o += sc[m] * Vs[m][d];
        out[(size_t)(win * NWIN + t) * CC + hoff + d] = (half_t)(o * inv);
    }
}

// ---------------------------------------------------------------------------
// host
// ---------------------------------------------------------------------------
extern "C" void kernel_launch(void* const* d_in, const int* in_sizes, int n_in,
                              void* d_out, int out_size, void* d_ws, size_t ws_size,
                              hipStream_t stream) {
    const float* x0 = (const float*)d_in[0];
    auto P = [&](int i) { return (const float*)d_in[i]; };

    // workspace layout
    char* ws = (char*)d_ws;
    const size_t SZ_F32  = (size_t)MROWS * CC * 4;       // 102.8 MB
    const size_t SZ_BIG  = (size_t)MROWS * HIDDEN_ * 2;  // 205.5 MB (qkv / h1 / mergedLN)
    const size_t SZ_ACT  = (size_t)MROWS * CC * 2;       // 51.4 MB
    float*  f32A  = (float*)(ws);
    float*  f32B  = (float*)(ws + SZ_F32);
    half_t* big16 = (half_t*)(ws + 2 * SZ_F32);
    half_t* act16 = (half_t*)(ws + 2 * SZ_F32 + SZ_BIG);
    half_t* w16   = (half_t*)(ws + 2 * SZ_F32 + SZ_BIG + SZ_ACT);

    half_t *wqkv[2], *wproj[2], *wfc1[2], *wfc2[2], *wred;
    size_t wp = 0;
    for (int b = 0; b < 2; ++b) {
        wqkv[b] = w16 + wp; wp += 768 * 256;
        wproj[b] = w16 + wp; wp += 256 * 256;
        wfc1[b] = w16 + wp; wp += 1024 * 256;
        wfc2[b] = w16 + wp; wp += 256 * 1024;
    }
    wred = w16 + wp;

    // convert weights to f16 (small, L2 resident afterwards)
    for (int b = 0; b < 2; ++b) {
        int base = 1 + b * 13;
        cvt_f32_f16<<<(768 * 256 + 255) / 256, 256, 0, stream>>>(P(base + 2), wqkv[b], 768 * 256);
        cvt_f32_f16<<<(256 * 256 + 255) / 256, 256, 0, stream>>>(P(base + 5), wproj[b], 256 * 256);
        cvt_f32_f16<<<(1024 * 256 + 255) / 256, 256, 0, stream>>>(P(base + 9), wfc1[b], 1024 * 256);
        cvt_f32_f16<<<(256 * 1024 + 255) / 256, 256, 0, stream>>>(P(base + 11), wfc2[b], 256 * 1024);
    }
    cvt_f32_f16<<<(512 * 1024 + 255) / 256, 256, 0, stream>>>(P(29), wred, 512 * 1024);

    for (int b = 0; b < 2; ++b) {
        int base = 1 + b * 13;
        const float* xin = (b == 0) ? x0 : f32B;
        float* x2 = f32A;
        float* x3 = f32B;
        int shift = (b == 0) ? 0 : SHIFT_;
        int masked = (b == 0) ? 0 : 1;

        // 1. LN1 + shift + window partition -> act16 [MROWS, 256] f16
        ln_kernel<<<MROWS, 256, 0, stream>>>(xin, P(base + 0), P(base + 1), act16, 1, shift);
        // 2. QKV GEMM (+bias, q scale) -> big16 [MROWS, 768] f16
        gemm_wmma_kernel<EPI_QKV><<<(MROWS / 32) * (768 / 64) / 4, 128, 0, stream>>>(
            act16, wqkv[b], P(base + 3), MROWS, 768, 256, big16, nullptr, nullptr, 0);
        // 3. window attention -> act16 [MROWS, 256] f16
        attn_kernel<<<WINB * NHEADS, 64, 0, stream>>>(big16, P(base + 4), act16, masked);
        // 4. proj GEMM + window-reverse + residual -> x2 f32
        gemm_wmma_kernel<EPI_PROJ><<<(MROWS / 32) * (256 / 64) / 4, 128, 0, stream>>>(
            act16, wproj[b], P(base + 6), MROWS, 256, 256, nullptr, x2, xin, shift);
        // 5. LN2 -> act16 f16
        ln_kernel<<<MROWS, 256, 0, stream>>>(x2, P(base + 7), P(base + 8), act16, 0, 0);
        // 6. FC1 GEMM + exact GELU -> big16 [MROWS, 1024] f16
        gemm_wmma_kernel<EPI_GELU><<<(MROWS / 32) * (1024 / 64) / 4, 128, 0, stream>>>(
            act16, wfc1[b], P(base + 10), MROWS, 1024, 256, big16, nullptr, nullptr, 0);
        // 7. FC2 GEMM + residual -> x3 f32
        gemm_wmma_kernel<EPI_RES><<<(MROWS / 32) * (256 / 64) / 4, 128, 0, stream>>>(
            big16, wfc2[b], P(base + 12), MROWS, 256, 1024, nullptr, x3, x2, 0);
    }

    // patch merging: gather 2x2 concat + LN(1024) -> big16 [25088, 1024] f16
    merge_ln_kernel<<<MMERGE, 256, 0, stream>>>(f32B, P(27), P(28), big16);
    // reduction GEMM -> d_out [25088, 512] f32
    gemm_wmma_kernel<EPI_PLAIN><<<(MMERGE / 32) * (512 / 64) / 4, 128, 0, stream>>>(
        big16, wred, nullptr, MMERGE, 512, 1024, nullptr, (float*)d_out, nullptr, 0);

    (void)in_sizes; (void)n_in; (void)out_size; (void)ws_size;
}
